// Decoder_19670950216508
// MI455X (gfx1250) — matrix operand order repaired and checked by reading, not compiled, not run on hardware
//
#include <hip/hip_runtime.h>

// GRU autoregressive decoder for MI455X (gfx1250, wave32, WMMA).
// Each wave owns 16 batch rows and runs the full 30-step recurrence with
// v_wmma_f32_16x16x32_f16 (f32 accumulate). Weights live in LDS pre-swizzled
// into B-fragment layout; h'/y round-trip through per-wave LDS tiles to turn
// the WMMA C layout (lane=col) into the A layout (lane=row) for the next step.
// 4 waves/block, 150 KB LDS -> 2 blocks can co-reside per 320 KB WGP.

typedef __attribute__((ext_vector_type(16))) _Float16 v16h;
typedef __attribute__((ext_vector_type(8)))  _Float16 v8h;
typedef __attribute__((ext_vector_type(8)))  float    v8f;

#define BATCH 16384
#define TDIM  60
#define IDIM  32
#define HDIM  128
#define GDIM  384      // 3*H
#define STEPS 30
#define SEQL  30

#define WAVES 4
#define BLOCK_THREADS (WAVES * 32)               // 128
#define ROWS_PER_WAVE 16
#define ROWS_PER_BLOCK (WAVES * ROWS_PER_WAVE)   // 64
#define NBLOCKS (BATCH / ROWS_PER_BLOCK)         // 256

__device__ __forceinline__ float fast_sigmoid(float x) {
  return 1.0f / (1.0f + __expf(-x));
}
__device__ __forceinline__ float fast_tanh(float x) {
  float e = __expf(-2.0f * x);
  return (1.0f - e) / (1.0f + e);
}

// Load a 16x32 f16 A-fragment from a row-major LDS buffer.
// Lane L (L<16): row M=L, K = 0..15.
// Lane L+16:     row M=L, K = 16..31.
__device__ __forceinline__ v16h load_fragA(const _Float16* buf, int rowStride,
                                           int kBase, int lane) {
  int m = lane & 15, half = lane >> 4;
  const _Float16* p = buf + m * rowStride + kBase + half * 16;
  v8h lo = *(const v8h*)p;          // ds_load_b128
  v8h hi = *(const v8h*)(p + 8);    // ds_load_b128
  return __builtin_shufflevector(lo, hi, 0,1,2,3,4,5,6,7,8,9,10,11,12,13,14,15);
}

// B fragments are pre-swizzled: lane L's 16 contiguous f16 (32B) hold its
// column/K-half slice. One aligned 32B LDS read (2x ds_load_b128).
__device__ __forceinline__ v16h load_fragB(const _Float16* frag, int lane) {
  return *(const v16h*)(frag + lane * 16);
}

__device__ __forceinline__ v8f wmma16(v16h a, v16h b, v8f c) {
  return __builtin_amdgcn_wmma_f32_16x16x32_f16(false, a, false, b,
                                                (short)0, c, false, false);
}

__global__ __launch_bounds__(BLOCK_THREADS)
void gru_decoder_kernel(const float* __restrict__ gx,  const float* __restrict__ gh,
                        const float* __restrict__ gWih, const float* __restrict__ gWhh,
                        const float* __restrict__ gbih, const float* __restrict__ gbhh,
                        const float* __restrict__ gWout, const float* __restrict__ gbout,
                        float* __restrict__ out) {
  // Weight fragments (f16, B layout): W_ihT 24 tiles x K32; W_hhT 24 tiles x 4
  // K-chunks; W_outT 2 tiles x 4 K-chunks. 512 f16 (1 KB) per fragment.
  __shared__ __align__(32) _Float16 sWih[24 * 512];   // 24 KB
  __shared__ __align__(32) _Float16 sWhh[96 * 512];   // 96 KB
  __shared__ __align__(32) _Float16 sWout[8 * 512];   //  8 KB
  __shared__ __align__(32) _Float16 sH[WAVES][ROWS_PER_WAVE * HDIM]; // 16 KB
  __shared__ __align__(32) _Float16 sX[WAVES][ROWS_PER_WAVE * IDIM]; //  4 KB
  __shared__ float sBrz[256];
  __shared__ float sBin[128], sBhn[128];
  __shared__ float sBout[32];

  const int tid  = threadIdx.x;
  const int lane = tid & 31;
  const int wv   = tid >> 5;
  const int rowBase = blockIdx.x * ROWS_PER_BLOCK + wv * ROWS_PER_WAVE;

  // ---------------- Stage weights into LDS as swizzled B fragments ----------
  // B[k][n] = W[n][k].  Fragment addr: frag*512 + (half*16 + n%16)*16 + (k%16).
  for (int i = tid; i < GDIM * IDIM; i += BLOCK_THREADS) {   // W_ih (384,32)
    int n = i >> 5, k = i & 31;
    int tIdx = n >> 4, nl = n & 15, half = k >> 4, kk = k & 15;
    sWih[tIdx * 512 + (half * 16 + nl) * 16 + kk] = (_Float16)gWih[i];
  }
  for (int i = tid; i < GDIM * HDIM; i += BLOCK_THREADS) {   // W_hh (384,128)
    int n = i >> 7, k = i & 127;
    int tIdx = n >> 4, nl = n & 15, c = k >> 5, kl = k & 31;
    int half = kl >> 4, kk = kl & 15;
    sWhh[(tIdx * 4 + c) * 512 + (half * 16 + nl) * 16 + kk] = (_Float16)gWhh[i];
  }
  for (int i = tid; i < IDIM * HDIM; i += BLOCK_THREADS) {   // W_out (32,128)
    int n = i >> 7, k = i & 127;
    int tIdx = n >> 4, nl = n & 15, c = k >> 5, kl = k & 31;
    int half = kl >> 4, kk = kl & 15;
    sWout[(tIdx * 4 + c) * 512 + (half * 16 + nl) * 16 + kk] = (_Float16)gWout[i];
  }
  for (int i = tid; i < 256; i += BLOCK_THREADS)             // fused r,z biases
    sBrz[i] = gbih[i] + gbhh[i];
  if (tid < 128) { sBin[tid] = gbih[256 + tid]; sBhn[tid] = gbhh[256 + tid]; }
  if (tid < 32)  sBout[tid] = gbout[tid];

  // ---------------- Per-wave state: h0 and x0 = x[:, SEQLEN, :] -------------
  for (int e = lane; e < ROWS_PER_WAVE * HDIM; e += 32) {
    int r = e >> 7, c = e & 127;
    sH[wv][e] = (_Float16)gh[(size_t)(rowBase + r) * HDIM + c];
  }
  for (int e = lane; e < ROWS_PER_WAVE * IDIM; e += 32) {
    int r = e >> 5, c = e & 31;
    sX[wv][e] = (_Float16)gx[((size_t)(rowBase + r) * TDIM + SEQL) * IDIM + c];
  }

  // h kept in f32 C-layout registers for the z*h blend (lane = column).
  const int cn   = lane & 15;
  const int rOff = (lane >> 4) * 8;
  v8f hC[8];
  #pragma unroll
  for (int j = 0; j < 8; ++j)
    #pragma unroll
    for (int v = 0; v < 8; ++v)
      hC[j][v] = gh[(size_t)(rowBase + rOff + v) * HDIM + j * 16 + cn];

  __syncthreads();
  asm volatile("s_wait_dscnt 0" ::: "memory");

  v16h hA[4], xA;
  #pragma unroll
  for (int c = 0; c < 4; ++c) hA[c] = load_fragA(sH[wv], HDIM, c * 32, lane);
  xA = load_fragA(sX[wv], IDIM, 0, lane);

  // ---------------- 30 autoregressive steps (kept rolled) -------------------
  #pragma unroll 1
  for (int t = 0; t < STEPS; ++t) {
    // Gates: for each 16-wide H tile j, 4 independent WMMA chains (r,z,gi_n,gh_n).
    // B-fragment loads are grouped ahead of their WMMA triplet so a single
    // dscnt wait covers 6 ds_load_b128 and the 3 WMMAs issue back-to-back.
    #pragma unroll
    for (int j = 0; j < 8; ++j) {
      v8f cr = {}, cz = {}, cgin = {}, cghn = {};
      {
        v16h br = load_fragB(sWih + (j)      * 512, lane);
        v16h bz = load_fragB(sWih + (8 + j)  * 512, lane);
        v16h bn = load_fragB(sWih + (16 + j) * 512, lane);
        cr   = wmma16(xA, br, cr);
        cz   = wmma16(xA, bz, cz);
        cgin = wmma16(xA, bn, cgin);
      }
      #pragma unroll
      for (int c = 0; c < 4; ++c) {
        v16h br = load_fragB(sWhh + ((j)      * 4 + c) * 512, lane);
        v16h bz = load_fragB(sWhh + ((8 + j)  * 4 + c) * 512, lane);
        v16h bn = load_fragB(sWhh + ((16 + j) * 4 + c) * 512, lane);
        cr   = wmma16(hA[c], br, cr);
        cz   = wmma16(hA[c], bz, cz);
        cghn = wmma16(hA[c], bn, cghn);
      }
      const float brz_r = sBrz[j * 16 + cn];
      const float brz_z = sBrz[128 + j * 16 + cn];
      const float b_in  = sBin[j * 16 + cn];
      const float b_hn  = sBhn[j * 16 + cn];
      _Float16* hdst = &sH[wv][0];
      #pragma unroll
      for (int v = 0; v < 8; ++v) {
        float r  = fast_sigmoid(cr[v] + brz_r);
        float z  = fast_sigmoid(cz[v] + brz_z);
        float n  = fast_tanh((cgin[v] + b_in) + r * (cghn[v] + b_hn));
        float hn = (1.0f - z) * n + z * hC[j][v];
        hC[j][v] = hn;
        hdst[(rOff + v) * HDIM + j * 16 + cn] = (_Float16)hn;  // C->rowmajor
      }
    }
    asm volatile("s_wait_dscnt 0" ::: "memory");
    #pragma unroll
    for (int c = 0; c < 4; ++c) hA[c] = load_fragA(sH[wv], HDIM, c * 32, lane);

    // y = h' @ W_outT + b_out ; y feeds back as next x and goes to global out.
    #pragma unroll
    for (int m = 0; m < 2; ++m) {
      v8f cy = {};
      #pragma unroll
      for (int c = 0; c < 4; ++c)
        cy = wmma16(hA[c], load_fragB(sWout + (m * 4 + c) * 512, lane), cy);
      const float bo = sBout[m * 16 + cn];
      #pragma unroll
      for (int v = 0; v < 8; ++v) {
        float y = cy[v] + bo;
        out[((size_t)(rowBase + rOff + v) * STEPS + t) * IDIM + m * 16 + cn] = y;
        sX[wv][(rOff + v) * IDIM + m * 16 + cn] = (_Float16)y;
      }
    }
    asm volatile("s_wait_dscnt 0" ::: "memory");
    xA = load_fragA(sX[wv], IDIM, 0, lane);
  }
}

extern "C" void kernel_launch(void* const* d_in, const int* in_sizes, int n_in,
                              void* d_out, int out_size, void* d_ws, size_t ws_size,
                              hipStream_t stream) {
  const float* x    = (const float*)d_in[0];
  const float* h    = (const float*)d_in[1];
  const float* Wih  = (const float*)d_in[2];
  const float* Whh  = (const float*)d_in[3];
  const float* bih  = (const float*)d_in[4];
  const float* bhh  = (const float*)d_in[5];
  const float* Wout = (const float*)d_in[6];
  const float* bout = (const float*)d_in[7];
  float* o = (float*)d_out;
  gru_decoder_kernel<<<dim3(NBLOCKS), dim3(BLOCK_THREADS), 0, stream>>>(
      x, h, Wih, Whh, bih, bhh, Wout, bout, o);
}
